// AutonomousRouter_75058848464997
// MI455X (gfx1250) — matrix operand order repaired and checked by reading, not compiled, not run on hardware
//
#include <hip/hip_runtime.h>

// Problem constants (fixed by the reference setup)
#define B_   4
#define N_   2048
#define D_   2048
#define E_   8
#define CB_  512          // bottleneck width
#define CAP_ 640          // expert capacity
#define M_   (B_ * N_)    // 8192 token rows
#define NT_  (E_ * CB_)   // 4096 total GEMM columns

// GEMM tiling
#define BM 128
#define BN 64
#define BK 64
#define KT (D_ / BK)      // 32 K-iterations
#define APITCH 72   // bf16 elements per A row in LDS (64 + 8 pad, keeps 16B align)
#define BPITCH 72   // bf16 elements per B column in LDS

typedef __bf16 bf16_t;
typedef bf16_t v16bf __attribute__((ext_vector_type(16)));
typedef float  v8f   __attribute__((ext_vector_type(8)));

union Frag { v16bf v; uint4 u[2]; };

// Round-to-nearest-even fp32 -> bf16, result in bits [31:16] of a 32-bit reg.
__device__ __forceinline__ unsigned int bfbits_hi(float x) {
  unsigned int u = __float_as_uint(x);
  return u + 0x7FFFu + ((u >> 16) & 1u);
}
__device__ __forceinline__ unsigned int pack2bf(float x, float y) {
  return (bfbits_hi(x) >> 16) | (bfbits_hi(y) & 0xFFFF0000u);
}
__device__ __forceinline__ unsigned short bf_lo(float x) {
  return (unsigned short)(bfbits_hi(x) >> 16);
}

// ---------------------------------------------------------------------------
// Pre-pass A: tokens fp32 -> bf16 (flat copy). 8 elements per thread.
// ---------------------------------------------------------------------------
__global__ __launch_bounds__(256)
void convertA_kernel(const float* __restrict__ A, unsigned short* __restrict__ Abf) {
  int idx = blockIdx.x * blockDim.x + threadIdx.x;    // M_*D_/8 threads
  size_t off = (size_t)idx * 8;
  const float4 a = *reinterpret_cast<const float4*>(A + off);
  const float4 b = *reinterpret_cast<const float4*>(A + off + 4);
  uint4 o;
  o.x = pack2bf(a.x, a.y);
  o.y = pack2bf(a.z, a.w);
  o.z = pack2bf(b.x, b.y);
  o.w = pack2bf(b.z, b.w);
  *reinterpret_cast<uint4*>(Abf + off) = o;
}

// ---------------------------------------------------------------------------
// Pre-pass B: weights [E][D][CB] fp32 -> transposed bf16 [E][CB][D].
// 64x64 tile transpose through LDS per block.
// ---------------------------------------------------------------------------
__global__ __launch_bounds__(256)
void convertBt_kernel(const float* __restrict__ W, unsigned short* __restrict__ Bt) {
  __shared__ __align__(16) unsigned short tile[64 * 72];
  const int tid = threadIdx.x;
  const int c0 = blockIdx.x * 64;
  const int k0 = blockIdx.y * 64;
  const int e  = blockIdx.z;
  // load 64(k) x 64(c) fp32 coalesced, convert, scatter into tile[c][k]
#pragma unroll
  for (int j = 0; j < 4; ++j) {
    int id = tid + j * 256;              // 0..1023 float4s
    int k  = id >> 4;
    int cq = id & 15;
    const float4 wv = *reinterpret_cast<const float4*>(
        W + ((size_t)e * D_ + k0 + k) * CB_ + c0 + cq * 4);
    tile[(cq * 4 + 0) * 72 + k] = bf_lo(wv.x);
    tile[(cq * 4 + 1) * 72 + k] = bf_lo(wv.y);
    tile[(cq * 4 + 2) * 72 + k] = bf_lo(wv.z);
    tile[(cq * 4 + 3) * 72 + k] = bf_lo(wv.w);
  }
  __syncthreads();
  // write tile[c][k] rows as contiguous bf16 (uint4 = 8 elements)
#pragma unroll
  for (int j = 0; j < 2; ++j) {
    int id = tid + j * 256;              // 0..511 uint4s
    int c  = id >> 3;
    int kq = id & 7;
    uint4 v = *reinterpret_cast<const uint4*>(&tile[c * 72 + kq * 8]);
    *reinterpret_cast<uint4*>(
        &Bt[((size_t)e * CB_ + c0 + c) * D_ + k0 + kq * 8]) = v;
  }
}

// ---------------------------------------------------------------------------
// Kernel 1 (fast path): bf16 WMMA GEMM from pre-converted Abf [M][D] and
// pre-transposed Bt [E][CB][D]; fused squared-column-sum; double-buffered.
// Hot loop is pure b128 copies + WMMA: no conversion VALU.
// ---------------------------------------------------------------------------
__global__ __launch_bounds__(256)
void gemm_norm_bf16_kernel(const unsigned short* __restrict__ Abf,
                           const unsigned short* __restrict__ Bt,
                           float* __restrict__ norm2) {
  __shared__ __align__(16) unsigned short As[2][BM * APITCH];
  __shared__ __align__(16) unsigned short Bs[2][BN * BPITCH];
  __shared__ float rowAcc[BM];

  const int tid  = threadIdx.x;
  const int row0 = blockIdx.y * BM;
  const int col0 = blockIdx.x * BN;
  const int eExp = col0 >> 9;
  const int cBase = col0 & (CB_ - 1);
  const unsigned short* Btb = Bt + (size_t)(eExp * CB_ + cBase) * D_;

  const int wave = tid >> 5;
  const int lane = tid & 31;
  const int wm = wave >> 1;
  const int wn = wave & 1;
  const int h  = lane >> 4;
  const int lm = lane & 15;

  v8f acc[2][2] = {};
  if (tid < BM) rowAcc[tid] = 0.0f;

  // prologue: stage tile 0
#pragma unroll
  for (int j = 0; j < 4; ++j) {          // A: 128 rows x 64 K = 1024 uint4
    int id = tid + j * 256;
    int r  = id >> 3;
    int kq = id & 7;
    uint4 v = *reinterpret_cast<const uint4*>(&Abf[(size_t)(row0 + r) * D_ + kq * 8]);
    *reinterpret_cast<uint4*>(&As[0][r * APITCH + kq * 8]) = v;
  }
#pragma unroll
  for (int j = 0; j < 2; ++j) {          // B: 64 cols x 64 K = 512 uint4
    int id = tid + j * 256;
    int nn = id >> 3;
    int kq = id & 7;
    uint4 v = *reinterpret_cast<const uint4*>(&Btb[(size_t)nn * D_ + kq * 8]);
    *reinterpret_cast<uint4*>(&Bs[0][nn * BPITCH + kq * 8]) = v;
  }
  __syncthreads();

  for (int kt = 0; kt < KT; ++kt) {
    const int cur = kt & 1, nxt = cur ^ 1;
    const bool haveNext = (kt + 1 < KT);
    const int k1 = (kt + 1) * BK;

    // 1) issue next tile's global loads (overlap with WMMA below)
    uint4 aR[4], bR[2];
    if (haveNext) {
#pragma unroll
      for (int j = 0; j < 4; ++j) {
        int id = tid + j * 256;
        int r  = id >> 3;
        int kq = id & 7;
        aR[j] = *reinterpret_cast<const uint4*>(
            &Abf[(size_t)(row0 + r) * D_ + k1 + kq * 8]);
      }
#pragma unroll
      for (int j = 0; j < 2; ++j) {
        int id = tid + j * 256;
        int nn = id >> 3;
        int kq = id & 7;
        bR[j] = *reinterpret_cast<const uint4*>(
            &Btb[(size_t)nn * D_ + k1 + kq * 8]);
      }
      if (kt + 2 < KT) {                 // warm L2 two tiles ahead
        __builtin_prefetch(Abf + (size_t)(row0 + (tid >> 1)) * D_ + (kt + 2) * BK + (tid & 1) * 32, 0, 1);
        __builtin_prefetch(Btb + (size_t)(tid >> 2) * D_ + (kt + 2) * BK + (tid & 3) * 16, 0, 1);
      }
    }

    // 2) 8 WMMAs from current buffer (fragment layouts per ISA 7.12.2)
    const unsigned short* Ac = As[cur];
    const unsigned short* Bc = Bs[cur];
#pragma unroll
    for (int ks = 0; ks < 2; ++ks) {
      const int ko = ks * 32;
      Frag a0, a1, b0, b1;
      {
        int r = wm * 32 + lm;
        a0.u[0] = *reinterpret_cast<const uint4*>(&Ac[r * APITCH + ko + h * 8]);
        a0.u[1] = *reinterpret_cast<const uint4*>(&Ac[r * APITCH + ko + 16 + h * 8]);
        a1.u[0] = *reinterpret_cast<const uint4*>(&Ac[(r + 16) * APITCH + ko + h * 8]);
        a1.u[1] = *reinterpret_cast<const uint4*>(&Ac[(r + 16) * APITCH + ko + 16 + h * 8]);
        int n = wn * 32 + lm;
        b0.u[0] = *reinterpret_cast<const uint4*>(&Bc[n * BPITCH + ko + h * 8]);
        b0.u[1] = *reinterpret_cast<const uint4*>(&Bc[n * BPITCH + ko + 16 + h * 8]);
        b1.u[0] = *reinterpret_cast<const uint4*>(&Bc[(n + 16) * BPITCH + ko + h * 8]);
        b1.u[1] = *reinterpret_cast<const uint4*>(&Bc[(n + 16) * BPITCH + ko + 16 + h * 8]);
      }
      acc[0][0] = __builtin_amdgcn_wmma_f32_16x16x32_bf16(false, a0.v, false, b0.v, (short)0, acc[0][0], false, false);
      acc[0][1] = __builtin_amdgcn_wmma_f32_16x16x32_bf16(false, a0.v, false, b1.v, (short)0, acc[0][1], false, false);
      acc[1][0] = __builtin_amdgcn_wmma_f32_16x16x32_bf16(false, a1.v, false, b0.v, (short)0, acc[1][0], false, false);
      acc[1][1] = __builtin_amdgcn_wmma_f32_16x16x32_bf16(false, a1.v, false, b1.v, (short)0, acc[1][1], false, false);
    }

    // 3) store staged registers into the other buffer (pure b128 copies)
    if (haveNext) {
      unsigned short* An = As[nxt];
      unsigned short* Bn = Bs[nxt];
#pragma unroll
      for (int j = 0; j < 4; ++j) {
        int id = tid + j * 256;
        int r  = id >> 3;
        int kq = id & 7;
        *reinterpret_cast<uint4*>(&An[r * APITCH + kq * 8]) = aR[j];
      }
#pragma unroll
      for (int j = 0; j < 2; ++j) {
        int id = tid + j * 256;
        int nn = id >> 3;
        int kq = id & 7;
        *reinterpret_cast<uint4*>(&Bn[nn * BPITCH + kq * 8]) = bR[j];
      }
    }
    __syncthreads();
  }

  // Epilogue: square + reduce over this block's 64 columns.
#pragma unroll
  for (int rt = 0; rt < 2; ++rt) {
#pragma unroll
    for (int v = 0; v < 8; ++v) {
      float s = acc[rt][0][v] * acc[rt][0][v] + acc[rt][1][v] * acc[rt][1][v];
      s += __shfl_xor(s, 1);
      s += __shfl_xor(s, 2);
      s += __shfl_xor(s, 4);
      s += __shfl_xor(s, 8);
      if (lm == v) {
        int rl = wm * 32 + rt * 16 + v + 8 * h;
        atomicAdd(&rowAcc[rl], s);
      }
    }
  }
  __syncthreads();
  if (tid < BM) {
    atomicAdd(&norm2[(size_t)(row0 + tid) * E_ + eExp], rowAcc[tid]);
  }
}

// ---------------------------------------------------------------------------
// Kernel 1 (fallback path, small workspace): convert-in-loop variant.
// ---------------------------------------------------------------------------
__global__ __launch_bounds__(256)
void gemm_norm_conv_kernel(const float* __restrict__ A, const float* __restrict__ W,
                           float* __restrict__ norm2) {
  __shared__ __align__(16) unsigned short As[2][BM * APITCH];
  __shared__ __align__(16) unsigned short Bs[2][BN * BPITCH];
  __shared__ float rowAcc[BM];

  const int tid  = threadIdx.x;
  const int row0 = blockIdx.y * BM;
  const int col0 = blockIdx.x * BN;
  const int eExp = col0 >> 9;
  const int cBase = col0 & (CB_ - 1);
  const float* Wbase = W + (size_t)eExp * D_ * CB_;

  const int wave = tid >> 5;
  const int lane = tid & 31;
  const int wm = wave >> 1;
  const int wn = wave & 1;
  const int h  = lane >> 4;
  const int lm = lane & 15;

  v8f acc[2][2] = {};
  if (tid < BM) rowAcc[tid] = 0.0f;

#pragma unroll
  for (int j = 0; j < 8; ++j) {
    int id = tid + j * 256;
    int r  = id >> 4;
    int kq = id & 15;
    const float4 av = *reinterpret_cast<const float4*>(
        A + (size_t)(row0 + r) * D_ + kq * 4);
    *reinterpret_cast<uint2*>(&As[0][r * APITCH + kq * 4]) =
        make_uint2(pack2bf(av.x, av.y), pack2bf(av.z, av.w));
  }
#pragma unroll
  for (int j = 0; j < 4; ++j) {
    int id = tid + j * 256;
    int kk = id >> 4;
    int nq = id & 15;
    const float4 wv = *reinterpret_cast<const float4*>(
        Wbase + (size_t)kk * CB_ + cBase + nq * 4);
    Bs[0][(nq * 4 + 0) * BPITCH + kk] = bf_lo(wv.x);
    Bs[0][(nq * 4 + 1) * BPITCH + kk] = bf_lo(wv.y);
    Bs[0][(nq * 4 + 2) * BPITCH + kk] = bf_lo(wv.z);
    Bs[0][(nq * 4 + 3) * BPITCH + kk] = bf_lo(wv.w);
  }
  __syncthreads();

  for (int kt = 0; kt < KT; ++kt) {
    const int cur = kt & 1, nxt = cur ^ 1;
    const bool haveNext = (kt + 1 < KT);
    const int k1 = (kt + 1) * BK;
    float4 aReg[8], bReg[4];
    if (haveNext) {
#pragma unroll
      for (int j = 0; j < 8; ++j) {
        int id = tid + j * 256;
        int r  = id >> 4;
        int kq = id & 15;
        aReg[j] = *reinterpret_cast<const float4*>(
            A + (size_t)(row0 + r) * D_ + k1 + kq * 4);
      }
#pragma unroll
      for (int j = 0; j < 4; ++j) {
        int id = tid + j * 256;
        int kk = id >> 4;
        int nq = id & 15;
        bReg[j] = *reinterpret_cast<const float4*>(
            Wbase + (size_t)(k1 + kk) * CB_ + cBase + nq * 4);
      }
    }
    const unsigned short* Ac = As[cur];
    const unsigned short* Bc = Bs[cur];
#pragma unroll
    for (int ks = 0; ks < 2; ++ks) {
      const int ko = ks * 32;
      Frag a0, a1, b0, b1;
      {
        int r = wm * 32 + lm;
        a0.u[0] = *reinterpret_cast<const uint4*>(&Ac[r * APITCH + ko + h * 8]);
        a0.u[1] = *reinterpret_cast<const uint4*>(&Ac[r * APITCH + ko + 16 + h * 8]);
        a1.u[0] = *reinterpret_cast<const uint4*>(&Ac[(r + 16) * APITCH + ko + h * 8]);
        a1.u[1] = *reinterpret_cast<const uint4*>(&Ac[(r + 16) * APITCH + ko + 16 + h * 8]);
        int n = wn * 32 + lm;
        b0.u[0] = *reinterpret_cast<const uint4*>(&Bc[n * BPITCH + ko + h * 8]);
        b0.u[1] = *reinterpret_cast<const uint4*>(&Bc[n * BPITCH + ko + 16 + h * 8]);
        b1.u[0] = *reinterpret_cast<const uint4*>(&Bc[(n + 16) * BPITCH + ko + h * 8]);
        b1.u[1] = *reinterpret_cast<const uint4*>(&Bc[(n + 16) * BPITCH + ko + 16 + h * 8]);
      }
      acc[0][0] = __builtin_amdgcn_wmma_f32_16x16x32_bf16(false, a0.v, false, b0.v, (short)0, acc[0][0], false, false);
      acc[0][1] = __builtin_amdgcn_wmma_f32_16x16x32_bf16(false, a0.v, false, b1.v, (short)0, acc[0][1], false, false);
      acc[1][0] = __builtin_amdgcn_wmma_f32_16x16x32_bf16(false, a1.v, false, b0.v, (short)0, acc[1][0], false, false);
      acc[1][1] = __builtin_amdgcn_wmma_f32_16x16x32_bf16(false, a1.v, false, b1.v, (short)0, acc[1][1], false, false);
    }
    if (haveNext) {
      unsigned short* An = As[nxt];
      unsigned short* Bn = Bs[nxt];
#pragma unroll
      for (int j = 0; j < 8; ++j) {
        int id = tid + j * 256;
        int r  = id >> 4;
        int kq = id & 15;
        *reinterpret_cast<uint2*>(&An[r * APITCH + kq * 4]) =
            make_uint2(pack2bf(aReg[j].x, aReg[j].y), pack2bf(aReg[j].z, aReg[j].w));
      }
#pragma unroll
      for (int j = 0; j < 4; ++j) {
        int id = tid + j * 256;
        int kk = id >> 4;
        int nq = id & 15;
        Bn[(nq * 4 + 0) * BPITCH + kk] = bf_lo(bReg[j].x);
        Bn[(nq * 4 + 1) * BPITCH + kk] = bf_lo(bReg[j].y);
        Bn[(nq * 4 + 2) * BPITCH + kk] = bf_lo(bReg[j].z);
        Bn[(nq * 4 + 3) * BPITCH + kk] = bf_lo(bReg[j].w);
      }
    }
    __syncthreads();
  }
#pragma unroll
  for (int rt = 0; rt < 2; ++rt) {
#pragma unroll
    for (int v = 0; v < 8; ++v) {
      float s = acc[rt][0][v] * acc[rt][0][v] + acc[rt][1][v] * acc[rt][1][v];
      s += __shfl_xor(s, 1);
      s += __shfl_xor(s, 2);
      s += __shfl_xor(s, 4);
      s += __shfl_xor(s, 8);
      if (lm == v) {
        int rl = wm * 32 + rt * 16 + v + 8 * h;
        atomicAdd(&rowAcc[rl], s);
      }
    }
  }
  __syncthreads();
  if (tid < BM) {
    atomicAdd(&norm2[(size_t)(row0 + tid) * E_ + eExp], rowAcc[tid]);
  }
}

// ---------------------------------------------------------------------------
// Kernel 2: logits = sqrt(norm2); softmax over E=8; stable top-2 indices.
// ---------------------------------------------------------------------------
__global__ __launch_bounds__(256)
void softmax_top2_kernel(const float* __restrict__ norm2,
                         float* __restrict__ probs, int* __restrict__ choice) {
  int t = blockIdx.x * blockDim.x + threadIdx.x;
  if (t >= M_) return;
  float lg[E_];
#pragma unroll
  for (int e = 0; e < E_; ++e) lg[e] = sqrtf(norm2[t * E_ + e]);
  float mx = lg[0];
#pragma unroll
  for (int e = 1; e < E_; ++e) mx = fmaxf(mx, lg[e]);
  float p[E_]; float sum = 0.0f;
#pragma unroll
  for (int e = 0; e < E_; ++e) { p[e] = expf(lg[e] - mx); sum += p[e]; }
  float inv = 1.0f / sum;
#pragma unroll
  for (int e = 0; e < E_; ++e) probs[t * E_ + e] = p[e] * inv;

  int i1 = 0;
#pragma unroll
  for (int e = 1; e < E_; ++e) if (lg[e] > lg[i1]) i1 = e;
  int i2 = (i1 == 0) ? 1 : 0;
#pragma unroll
  for (int e = 0; e < E_; ++e) if (e != i1 && lg[e] > lg[i2]) i2 = e;

  int b = t >> 11, n = t & (N_ - 1);
  choice[(b * 2 + 0) * N_ + n] = i1;
  choice[(b * 2 + 1) * N_ + n] = i2;
}

// ---------------------------------------------------------------------------
// Kernel 3: running-count scan via ballot (k-major order). tprio preset to -1.
// ---------------------------------------------------------------------------
__global__ __launch_bounds__(256)
void prio_kernel(const int* __restrict__ choice, int* __restrict__ tprio) {
  int b = blockIdx.x;
  int e = threadIdx.x >> 5;
  int lane = threadIdx.x & 31;
  const int* ch = choice + (size_t)b * 2 * N_;
  int base = 0;
  for (int p0 = 0; p0 < 2 * N_; p0 += 32) {
    int p = p0 + lane;
    int c = ch[p];
    bool match = (c == e);
    unsigned long long m = __ballot(match);
    int before = __popcll(m & ((1ull << lane) - 1ull));
    if (match) {
      int n = p & (N_ - 1);
      tprio[((size_t)b * N_ + n) * E_ + e] = base + before;
    }
    base += __popcll(m);
  }
}

// ---------------------------------------------------------------------------
// Kernel 4: stream [2,B,N,E,CAP] one-hot output (coalesced b128 stores).
// ---------------------------------------------------------------------------
__global__ __launch_bounds__(256)
void out_kernel(const float* __restrict__ probs, const int* __restrict__ tprio,
                float* __restrict__ out) {
  int idx = blockIdx.x * blockDim.x + threadIdx.x;
  int c = (idx % (CAP_ / 4)) * 4;
  int rest = idx / (CAP_ / 4);
  int e = rest % E_;   rest /= E_;
  int n = rest % N_;   rest /= N_;
  int b = rest % B_;
  int s = rest / B_;
  int row = ((b * N_ + n) * E_) + e;
  int tp = tprio[row];
  float val = (s == 0) ? 1.0f : probs[row];
  float4 o = make_float4(0.f, 0.f, 0.f, 0.f);
  if (tp >= c && tp < c + 4) reinterpret_cast<float*>(&o)[tp - c] = val;
  reinterpret_cast<float4*>(out)[idx] = o;
}

// ---------------------------------------------------------------------------
extern "C" void kernel_launch(void* const* d_in, const int* in_sizes, int n_in,
                              void* d_out, int out_size, void* d_ws, size_t ws_size,
                              hipStream_t stream) {
  const float* tok = (const float*)d_in[0];   // [B,N,D] fp32
  const float* wgt = (const float*)d_in[1];   // [E,D,CB] fp32

  char* ws = (char*)d_ws;
  float* norm2 = (float*)(ws);                          // 65536 floats
  float* probs = (float*)(ws + 262144);                 // 65536 floats
  int*   choice = (int*)(ws + 524288);                  // 16384 ints
  int*   tprio  = (int*)(ws + 589824);                  // 65536 ints
  unsigned short* Abf = (unsigned short*)(ws + (1 << 20));              // 32 MB
  unsigned short* Bt  = (unsigned short*)(ws + (1 << 20) + (size_t)M_ * D_ * 2);  // 16 MB
  const size_t NEED = (1 << 20) + (size_t)M_ * D_ * 2 + (size_t)E_ * D_ * CB_ * 2;

  hipMemsetAsync(norm2, 0x00, (size_t)M_ * E_ * sizeof(float), stream);
  hipMemsetAsync(tprio, 0xFF, (size_t)M_ * E_ * sizeof(int), stream);   // -1

  dim3 g1(NT_ / BN, M_ / BM);                           // 64 x 64 blocks
  if (ws_size >= NEED) {
    convertA_kernel<<<(M_ * D_ / 8) / 256, 256, 0, stream>>>(tok, Abf);
    convertBt_kernel<<<dim3(CB_ / 64, D_ / 64, E_), 256, 0, stream>>>(wgt, Bt);
    gemm_norm_bf16_kernel<<<g1, 256, 0, stream>>>(Abf, Bt, norm2);
  } else {
    gemm_norm_conv_kernel<<<g1, 256, 0, stream>>>(tok, wgt, norm2);
  }

  softmax_top2_kernel<<<(M_ + 255) / 256, 256, 0, stream>>>(norm2, probs, choice);

  prio_kernel<<<B_, 256, 0, stream>>>(choice, tprio);

  int total4 = 2 * B_ * N_ * E_ * (CAP_ / 4);           // 20,971,520
  out_kernel<<<total4 / 256, 256, 0, stream>>>(probs, tprio, (float*)d_out);
}
